// PathConv_20486994002223
// MI455X (gfx1250) — matrix operand order repaired and checked by reading, not compiled, not run on hardware
//
#include <hip/hip_runtime.h>
#include <hip/hip_bf16.h>

typedef __attribute__((ext_vector_type(16))) _Float16 v16h;
typedef __attribute__((ext_vector_type(8)))  _Float16 v8h;
typedef __attribute__((ext_vector_type(8)))  float    v8f;

#define D_OUT 256
#define D_CELL 64

union FragU { v16h v; v8h h8[2]; _Float16 e[16]; };
union AccU  { v8f v; float f[8]; };

// ---------------- Phase 1: edge softmax + CSR aggregation ----------------

__global__ void k_init_nodes(unsigned int* __restrict__ mbits,
                             float* __restrict__ ssum,
                             int* __restrict__ deg, int N) {
    int i = blockIdx.x * blockDim.x + threadIdx.x;
    if (i < N) { mbits[i] = 0u; ssum[i] = 0.0f; deg[i] = 0; }
}

// one wave per edge: e = relu(dot(h[src], attn)); atomicMax(m); deg count
__global__ void k_edge_attn(const float* __restrict__ h,
                            const float* __restrict__ attn,
                            const int* __restrict__ src,
                            const int* __restrict__ dst,
                            float* __restrict__ eA,
                            unsigned int* __restrict__ mbits,
                            int* __restrict__ deg, int E) {
    int wid  = (int)((blockIdx.x * blockDim.x + threadIdx.x) >> 5);
    int lane = threadIdx.x & 31;
    if (wid >= E) return;
    int s = src[wid];
    const float4* hp = (const float4*)(h + (size_t)s * D_OUT + lane * 8);
    const float4* ap = (const float4*)(attn + lane * 8);
    float4 h0 = hp[0], h1 = hp[1];
    float4 a0 = ap[0], a1 = ap[1];
    float p = h0.x*a0.x + h0.y*a0.y + h0.z*a0.z + h0.w*a0.w
            + h1.x*a1.x + h1.y*a1.y + h1.z*a1.z + h1.w*a1.w;
    #pragma unroll
    for (int o = 16; o > 0; o >>= 1) p += __shfl_down(p, o, 32);
    if (lane == 0) {
        float ev = p > 0.0f ? p : 0.0f;   // relu -> ev >= 0: uint-bit atomicMax is order-preserving
        int d = dst[wid];
        eA[wid] = ev;
        atomicMax(mbits + d, __float_as_uint(ev));
        atomicAdd(deg + d, 1);
    }
}

// single-block exclusive scan of deg -> off, cursor; off[N] = E
__global__ void k_scan(const int* __restrict__ deg,
                       int* __restrict__ off,
                       int* __restrict__ cur, int N) {
    __shared__ int sh[1024];
    int tid = threadIdx.x;
    int carry = 0;
    for (int base = 0; base < N; base += 1024) {
        int i = base + tid;
        int val = (i < N) ? deg[i] : 0;
        sh[tid] = val;
        __syncthreads();
        for (int ofs = 1; ofs < 1024; ofs <<= 1) {
            int t = (tid >= ofs) ? sh[tid - ofs] : 0;
            __syncthreads();
            sh[tid] += t;
            __syncthreads();
        }
        int excl = sh[tid] - val + carry;
        if (i < N) { off[i] = excl; cur[i] = excl; }
        __syncthreads();
        carry += sh[1023];
        __syncthreads();
    }
    if (tid == 0) off[N] = carry;
}

// thread per edge: ex = exp(e - m[dst]); s += ex; place edge id into CSR slot
__global__ void k_edge_exp(float* __restrict__ eA,
                           const unsigned int* __restrict__ mbits,
                           float* __restrict__ ssum,
                           const int* __restrict__ dst,
                           int* __restrict__ cur,
                           int* __restrict__ perm, int E) {
    int i = blockIdx.x * blockDim.x + threadIdx.x;
    if (i >= E) return;
    int d = dst[i];
    float m  = __uint_as_float(mbits[d]);
    float ex = __expf(eA[i] - m);
    eA[i] = ex;
    atomicAdd(ssum + d, ex);
    int pos = atomicAdd(cur + d, 1);
    perm[pos] = i;
}

// thread per edge: a = ex / s[dst]  (in place)
__global__ void k_edge_norm(float* __restrict__ eA,
                            const float* __restrict__ ssum,
                            const int* __restrict__ dst, int E) {
    int i = blockIdx.x * blockDim.x + threadIdx.x;
    if (i >= E) return;
    eA[i] = eA[i] / ssum[dst[i]];
}

// one wave per dst node: h_neigh1[v] = sum_{edges->v} a * h[src], store f16
__global__ void k_aggregate(const float* __restrict__ h,
                            const float* __restrict__ aA,
                            const int* __restrict__ src,
                            const int* __restrict__ off,
                            const int* __restrict__ perm,
                            _Float16* __restrict__ hn16, int N) {
    int wid  = (int)((blockIdx.x * blockDim.x + threadIdx.x) >> 5);
    int lane = threadIdx.x & 31;
    if (wid >= N) return;
    int beg = off[wid], end = off[wid + 1];
    float acc[8];
    #pragma unroll
    for (int j = 0; j < 8; j++) acc[j] = 0.0f;
    for (int j = beg; j < end; j++) {
        int   eidx = perm[j];
        int   sv   = src[eidx];
        float w    = aA[eidx];
        const float4* hp = (const float4*)(h + (size_t)sv * D_OUT + lane * 8);
        float4 x0 = hp[0], x1 = hp[1];
        acc[0] += x0.x * w; acc[1] += x0.y * w; acc[2] += x0.z * w; acc[3] += x0.w * w;
        acc[4] += x1.x * w; acc[5] += x1.y * w; acc[6] += x1.z * w; acc[7] += x1.w * w;
    }
    union { v8h v; _Float16 e[8]; } o;
    #pragma unroll
    for (int j = 0; j < 8; j++) o.e[j] = (_Float16)acc[j];
    *(v8h*)(hn16 + (size_t)wid * D_OUT + lane * 8) = o.v;
}

// ---------------- Weight packing: f32 [K x 256] -> f16 B-fragments ----------------
// Fragment (kt, ct): K rows [kt*32, kt*32+32) x cols [ct*16, ct*16+16).
// Per-lane layout (WMMA B operand): lane l holds column ct*16 + (l&15);
// elements j=0..15 hold K = kt*32 + (l>>4)*16 + j.
// Packed half-index: ((kt*16 + ct)*32 + lane)*16 + j

__global__ void k_pack_w(const float* __restrict__ W, _Float16* __restrict__ Wp, int K) {
    int t = blockIdx.x * blockDim.x + threadIdx.x;
    if (t >= K * 256) return;
    int frag   = t >> 9;
    int within = t & 511;
    int lane   = within >> 4;
    int j      = within & 15;
    int kt     = frag >> 4;      // 16 col-tiles per k-tile row (256/16)
    int ct     = frag & 15;
    int k      = kt * 32 + (lane >> 4) * 16 + j;
    int col    = ct * 16 + (lane & 15);
    Wp[t] = (_Float16)W[(size_t)k * 256 + col];
}

__global__ void k_f32_to_f16(const float* __restrict__ x, _Float16* __restrict__ y, int n) {
    int i = blockIdx.x * blockDim.x + threadIdx.x;
    if (i < n) y[i] = (_Float16)x[i];
}

// ---------------- Phase 2: WMMA GEMMs ----------------
// Block = 256 threads = 8 waves. Workgroup tile: 128 rows x 128 cols.
// Wave: 16 rows x 8 col-tiles; software-pipelined with a 4-deep B-fragment
// rotation (prefetch distance 2) and sched_barriers to pin the pipeline.
// Ab: per-lane A base; frag kt at half-offsets kt*32 / kt*32+8
// Bb: per-lane packed-B base; frag f=kt*8+c at half-offset f*512

__device__ __forceinline__ void load_afrag(FragU& a, const _Float16* Ab, int kt) {
    a.h8[0] = *(const v8h*)(Ab + kt * 32);
    a.h8[1] = *(const v8h*)(Ab + kt * 32 + 8);
}

__device__ __forceinline__ void load_bfrag(FragU& b, const _Float16* Bb, int f) {
    b.h8[0] = *(const v8h*)(Bb + f * 512);
    b.h8[1] = *(const v8h*)(Bb + f * 512 + 8);
}

template <int K>
__device__ __forceinline__ void gemm_accum(const _Float16* __restrict__ Ab,
                                           const _Float16* __restrict__ Bb,
                                           v8f acc[8]) {
    constexpr int NKT = K / 32;
    constexpr int NF  = NKT * 8;
    FragU a[2], b[4];
    load_afrag(a[0], Ab, 0);
    load_bfrag(b[0], Bb, 0);
    load_bfrag(b[1], Bb, 1);
    #pragma unroll
    for (int kt = 0; kt < NKT; kt++) {
        #pragma unroll
        for (int c = 0; c < 8; c++) {
            const int f = kt * 8 + c;
            if (c == 0 && kt + 1 < NKT) load_afrag(a[(kt + 1) & 1], Ab, kt + 1);
            if (f + 2 < NF)             load_bfrag(b[(f + 2) & 3], Bb, f + 2);
            // keep the prefetch loads above the consuming WMMA
            __builtin_amdgcn_sched_barrier(0);
            acc[c] = __builtin_amdgcn_wmma_f32_16x16x32_f16(
                false, a[kt & 1].v, false, b[f & 3].v, (short)0, acc[c], false, false);
        }
    }
}

template <int K>
__global__ __launch_bounds__(256)
void k_gemm_relu_h16(const _Float16* __restrict__ A,
                     const _Float16* __restrict__ Wp,
                     const float* __restrict__ bias,
                     _Float16* __restrict__ C, int M) {
    int lane   = threadIdx.x & 31;
    int wave   = threadIdx.x >> 5;
    int mbase  = blockIdx.x * 128 + wave * 16;
    int ctbase = blockIdx.y * 8;
    int kh     = lane >> 4;
    int r      = mbase + (lane & 15); if (r >= M) r = M - 1;

    const _Float16* Ab = A  + (size_t)r * K + kh * 16;
    const _Float16* Bb = Wp + ((size_t)ctbase * 32 + lane) * 16;

    v8f acc[8] = {};
    gemm_accum<K>(Ab, Bb, acc);

    int rb = mbase + kh * 8;
    #pragma unroll
    for (int c = 0; c < 8; c++) {
        int col = (ctbase + c) * 16 + (lane & 15);
        AccU u; u.v = acc[c];
        float bv = bias[col];
        #pragma unroll
        for (int i = 0; i < 8; i++) {
            int rr = rb + i;
            if (rr < M) {
                float val = u.f[i] + bv;
                C[(size_t)rr * D_OUT + col] = (_Float16)(val > 0.0f ? val : 0.0f);
            }
        }
    }
}

// out = relu(A1 @ W1 + A2 @ W2 + b1 + b2), f32 output. K = 256 for both.
__global__ __launch_bounds__(256)
void k_gemm2_out(const _Float16* __restrict__ A1, const _Float16* __restrict__ W1p,
                 const _Float16* __restrict__ A2, const _Float16* __restrict__ W2p,
                 const float* __restrict__ b1, const float* __restrict__ b2,
                 int M, float* __restrict__ Out) {
    int lane   = threadIdx.x & 31;
    int wave   = threadIdx.x >> 5;
    int mbase  = blockIdx.x * 128 + wave * 16;
    int ctbase = blockIdx.y * 8;
    int kh     = lane >> 4;
    int r      = mbase + (lane & 15); if (r >= M) r = M - 1;

    size_t boff = ((size_t)ctbase * 32 + lane) * 16;
    const _Float16* A1b = A1 + (size_t)r * D_OUT + kh * 16;
    const _Float16* A2b = A2 + (size_t)r * D_OUT + kh * 16;

    v8f acc[8] = {};
    gemm_accum<D_OUT>(A1b, W1p + boff, acc);
    gemm_accum<D_OUT>(A2b, W2p + boff, acc);

    int rb = mbase + kh * 8;
    #pragma unroll
    for (int c = 0; c < 8; c++) {
        int col = (ctbase + c) * 16 + (lane & 15);
        AccU u; u.v = acc[c];
        float bv = b1[col] + b2[col];
        #pragma unroll
        for (int i = 0; i < 8; i++) {
            int rr = rb + i;
            if (rr < M) {
                float val = u.f[i] + bv;
                Out[(size_t)rr * D_OUT + col] = val > 0.0f ? val : 0.0f;
            }
        }
    }
}

// ---------------- host ----------------

extern "C" void kernel_launch(void* const* d_in, const int* in_sizes, int n_in,
                              void* d_out, int out_size, void* d_ws, size_t ws_size,
                              hipStream_t stream) {
    const float* h    = (const float*)d_in[0];
    const float* cell = (const float*)d_in[1];
    const float* attn = (const float*)d_in[2];
    const float* Wn1  = (const float*)d_in[3];
    const float* bn1  = (const float*)d_in[4];
    const float* Wn2  = (const float*)d_in[5];
    const float* bn2  = (const float*)d_in[6];
    const float* Ws1  = (const float*)d_in[7];
    const float* bs1  = (const float*)d_in[8];
    const float* Ws2  = (const float*)d_in[9];
    const float* bs2  = (const float*)d_in[10];
    const int*   src  = (const int*)d_in[11];
    const int*   dst  = (const int*)d_in[12];

    int N = in_sizes[0] / D_OUT;   // 50000
    int E = in_sizes[11];          // 800000

    // carve workspace (256B aligned)
    char* w = (char*)d_ws;
    auto alloc = [&](size_t bytes) -> void* {
        void* p = (void*)w;
        w += (bytes + 255) & ~(size_t)255;
        return p;
    };
    float*        eA    = (float*)alloc((size_t)E * 4);        // e -> ex -> a (in place)
    unsigned int* mbits = (unsigned int*)alloc((size_t)N * 4);
    float*        ssum  = (float*)alloc((size_t)N * 4);
    int*          deg   = (int*)alloc((size_t)N * 4);
    int*          offs  = (int*)alloc((size_t)(N + 1) * 4);
    int*          cur   = (int*)alloc((size_t)N * 4);
    int*          perm  = (int*)alloc((size_t)E * 4);
    _Float16*     hn16  = (_Float16*)alloc((size_t)N * D_OUT * 2);  // h_neigh1 (f16)
    _Float16*     hs16  = (_Float16*)alloc((size_t)N * D_OUT * 2);  // hidden(self)
    _Float16*     hh16  = (_Float16*)alloc((size_t)N * D_OUT * 2);  // hidden(neigh)
    _Float16*     cf16  = (_Float16*)alloc((size_t)N * D_CELL * 2); // cell_feat (f16)
    _Float16*     Ws1p  = (_Float16*)alloc((size_t)D_CELL * 256 * 2);
    _Float16*     Wn1p  = (_Float16*)alloc((size_t)D_OUT * 256 * 2);
    _Float16*     Ws2p  = (_Float16*)alloc((size_t)D_OUT * 256 * 2);
    _Float16*     Wn2p  = (_Float16*)alloc((size_t)D_OUT * 256 * 2);

    int Mt = (N + 127) / 128;

    // phase 1
    k_init_nodes<<<(N + 255) / 256, 256, 0, stream>>>(mbits, ssum, deg, N);
    k_edge_attn <<<(E + 7) / 8, 256, 0, stream>>>(h, attn, src, dst, eA, mbits, deg, E);
    k_scan      <<<1, 1024, 0, stream>>>(deg, offs, cur, N);
    k_edge_exp  <<<(E + 255) / 256, 256, 0, stream>>>(eA, mbits, ssum, dst, cur, perm, E);
    k_edge_norm <<<(E + 255) / 256, 256, 0, stream>>>(eA, ssum, dst, E);
    k_aggregate <<<(N + 7) / 8, 256, 0, stream>>>(h, eA, src, offs, perm, hn16, N);

    // weight packing + activation convert (~1 MB total, one-shot)
    k_pack_w<<<(D_CELL * 256 + 255) / 256, 256, 0, stream>>>(Ws1, Ws1p, D_CELL);
    k_pack_w<<<(D_OUT  * 256 + 255) / 256, 256, 0, stream>>>(Wn1, Wn1p, D_OUT);
    k_pack_w<<<(D_OUT  * 256 + 255) / 256, 256, 0, stream>>>(Ws2, Ws2p, D_OUT);
    k_pack_w<<<(D_OUT  * 256 + 255) / 256, 256, 0, stream>>>(Wn2, Wn2p, D_OUT);
    k_f32_to_f16<<<(N * D_CELL + 255) / 256, 256, 0, stream>>>(cell, cf16, N * D_CELL);

    // phase 2
    dim3 ggrid(Mt, D_OUT / 128);
    // hidden(self)  = relu(cell_feat @ Ws1 + bs1)   (K = 64)
    k_gemm_relu_h16<D_CELL><<<ggrid, 256, 0, stream>>>(cf16, Ws1p, bs1, hs16, N);
    // hidden(neigh) = relu(h_neigh1 @ Wn1 + bn1)    (K = 256)
    k_gemm_relu_h16<D_OUT><<<ggrid, 256, 0, stream>>>(hn16, Wn1p, bn1, hh16, N);
    // out = relu(hidden_self @ Ws2 + hidden_neigh @ Wn2 + bs2 + bn2)
    k_gemm2_out<<<ggrid, 256, 0, stream>>>(hs16, Ws2p, hh16, Wn2p, bs2, bn2, N, (float*)d_out);
}
